// SE3TransformerLayer_65644280152627
// MI455X (gfx1250) — matrix operand order repaired and checked by reading, not compiled, not run on hardware
//
#include <hip/hip_runtime.h>
#include <hip/hip_bf16.h>

// ---------------------------------------------------------------------------
// SE(3) transformer layer for MI455X (gfx1250), wave32 + WMMA f16 (f32 accum)
// ---------------------------------------------------------------------------

#define BATCH 2
#define SEQ   2048
#define HID   768
#define NH    8
#define HD    96
#define DVC   256
#define TOK   (BATCH * SEQ)          // 4096

typedef __attribute__((ext_vector_type(16))) _Float16 v16h;
typedef __attribute__((ext_vector_type(8)))  float    v8f;

// ---------------- WMMA fragment helpers (layouts per CDNA5 ISA) --

// A-matrix 16x32 f16: lane = hf*16 + row; row = lane&15.
// v16h elems 0..7  -> K = hf*8 + e        (VGPR0..3)
// v16h elems 8..15 -> K = 16 + hf*8 + e   (VGPR4..7)
__device__ __forceinline__ v16h load_fragA_h(const _Float16* base, int ld) {
    const int lane = threadIdx.x & 31;
    const int row  = lane & 15;
    const int hf   = lane >> 4;
    const _Float16* p = base + row * ld;
    v16h f;
#pragma unroll
    for (int e = 0; e < 8; ++e) f[e]     = p[hf * 8 + e];
#pragma unroll
    for (int e = 0; e < 8; ++e) f[8 + e] = p[16 + hf * 8 + e];
    return f;
}

// Same A layout but source is f32 (with per-row scale folded in).
__device__ __forceinline__ v16h load_fragA_f(const float* base, int ld, float scale) {
    const int lane = threadIdx.x & 31;
    const int row  = lane & 15;
    const int hf   = lane >> 4;
    const float* p = base + row * ld;
    v16h f;
#pragma unroll
    for (int e = 0; e < 8; ++e) f[e]     = (_Float16)(p[hf * 8 + e] * scale);
#pragma unroll
    for (int e = 0; e < 8; ++e) f[8 + e] = (_Float16)(p[16 + hf * 8 + e] * scale);
    return f;
}

// B-matrix 32x16 f16 (contraction K x columns N), stored transposed in memory
// as [n][k] with row stride ld (contraction contiguous).  Same K order as A:
// elems 0..7 -> K = hf*8 + e, elems 8..15 -> K = 16 + hf*8 + e; n = lane&15.
__device__ __forceinline__ v16h load_fragB_h(const _Float16* base, int ld) {
    const int lane = threadIdx.x & 31;
    const int n    = lane & 15;
    const int hf   = lane >> 4;
    const _Float16* p = base + n * ld;
    v16h f;
#pragma unroll
    for (int e = 0; e < 8; ++e) f[e]     = p[hf * 8 + e];
#pragma unroll
    for (int e = 0; e < 8; ++e) f[8 + e] = p[16 + hf * 8 + e];
    return f;
}

__device__ __forceinline__ v8f wmma16(v16h a, v16h b, v8f c) {
    return __builtin_amdgcn_wmma_f32_16x16x32_f16(
        /*neg_a=*/false, a, /*neg_b=*/false, b,
        /*c_mod=*/(short)0, c, /*reuse_a=*/false, /*reuse_b=*/false);
}

// C/D 16x16 f32: VGPR j, lanes0-15 -> M=j, lanes16-31 -> M=j+8; N = lane&15.

// ---------------- conversion kernels ---------------------------------------

__global__ void __launch_bounds__(256) cvt_f32_f16_kernel(
    const float* __restrict__ src, _Float16* __restrict__ dst, int n) {
    int i = blockIdx.x * 256 + threadIdx.x;
    if (i < n) dst[i] = (_Float16)src[i];
}

// vector_features [b][s][d][c] (f32) -> planes [b][c][d][s] (f16)
__global__ void __launch_bounds__(256) cvt_vecf_kernel(
    const float* __restrict__ src, _Float16* __restrict__ dst) {
    int i = blockIdx.x * 256 + threadIdx.x;          // < B*S*DVC*3
    int c = i % 3;
    int d = (i / 3) & (DVC - 1);
    int s = (i / (3 * DVC)) & (SEQ - 1);
    int b = i / (3 * DVC * SEQ);
    dst[((b * 3 + c) * DVC + d) * SEQ + s] = (_Float16)src[i];
}

// ---------------- fused QKV projection -------------------------------------
// q,k: [token][HID] f16 ; v: transposed [b*HID + chan][s] f16
__global__ void __launch_bounds__(256) gemm_qkv_kernel(
    const _Float16* __restrict__ xh,
    const _Float16* __restrict__ wq, const _Float16* __restrict__ wk,
    const _Float16* __restrict__ wv,
    const float* __restrict__ bq, const float* __restrict__ bk,
    const float* __restrict__ bv,
    _Float16* __restrict__ qo, _Float16* __restrict__ ko,
    _Float16* __restrict__ vT) {
    const int z = blockIdx.z;
    const _Float16* W    = (z == 0) ? wq : (z == 1) ? wk : wv;
    const float*    bias = (z == 0) ? bq : (z == 1) ? bk : bv;

    const int wave = threadIdx.x >> 5;
    const int gw   = blockIdx.x * 8 + wave;           // 3072 waves
    const int mt   = gw % (TOK / 16);
    const int nb   = gw / (TOK / 16);                 // 0..11
    const int m0   = mt * 16, n0 = nb * 64;

    v8f acc[4] = {};
#pragma unroll 4
    for (int ks = 0; ks < HID / 32; ++ks) {
        v16h a = load_fragA_h(xh + m0 * HID + ks * 32, HID);
#pragma unroll
        for (int t = 0; t < 4; ++t) {
            v16h b = load_fragB_h(W + (n0 + t * 16) * HID + ks * 32, HID);
            acc[t] = wmma16(a, b, acc[t]);
        }
    }
    const int lane = threadIdx.x & 31, hf = lane >> 4, nl = lane & 15;
#pragma unroll
    for (int t = 0; t < 4; ++t) {
#pragma unroll
        for (int j = 0; j < 8; ++j) {
            int m = m0 + j + 8 * hf;
            int n = n0 + t * 16 + nl;
            float v = acc[t][j] + bias[n];
            if (z == 0)      qo[m * HID + n] = (_Float16)v;
            else if (z == 1) ko[m * HID + n] = (_Float16)v;
            else {
                int b_ = m >> 11, s_ = m & (SEQ - 1);
                vT[(b_ * HID + n) * SEQ + s_] = (_Float16)v;   // (b*NH+h)*HD+hd == b*768+n
            }
        }
    }
}

// ---------------- attention core -------------------------------------------
// 1 wave per (b, head, 16-query tile).  128KB dynamic LDS = f32 score block.
__global__ void __launch_bounds__(32) attn1_kernel(
    const _Float16* __restrict__ qh, const _Float16* __restrict__ kh,
    const _Float16* __restrict__ vT,
    float* __restrict__ attn_mean, _Float16* __restrict__ ash) {
    extern __shared__ float P[];                      // [16][SEQ]
    const int wg = blockIdx.x;
    const int qt = wg & 127;
    const int h  = (wg >> 7) & 7;
    const int b  = wg >> 10;
    const int q0 = qt * 16;
    const int lane = threadIdx.x;
    const int hf = lane >> 4, nl = lane & 15;
    const float rscale = 0.1020620726f;               // 1/sqrt(96)

    // Q fragments (contraction = head dim, contiguous)
    const _Float16* Qb = qh + ((b * SEQ + q0) * HID + h * HD);
    v16h aq0 = load_fragA_h(Qb,      HID);
    v16h aq1 = load_fragA_h(Qb + 32, HID);
    v16h aq2 = load_fragA_h(Qb + 64, HID);

    // ---- scores: S = Q Kt / sqrt(hd) into LDS --------------------------
    for (int kt = 0; kt < SEQ / 16; ++kt) {
        const _Float16* Kb = kh + ((b * SEQ + kt * 16) * HID + h * HD);
        v8f c = {};
        c = wmma16(aq0, load_fragB_h(Kb,      HID), c);
        c = wmma16(aq1, load_fragB_h(Kb + 32, HID), c);
        c = wmma16(aq2, load_fragB_h(Kb + 64, HID), c);
#pragma unroll
        for (int j = 0; j < 8; ++j)
            P[(j + 8 * hf) * SEQ + kt * 16 + nl] = c[j] * rscale;
    }
    __syncthreads();

    // ---- exact two-pass softmax; lane pair (L, L+16) shares row L&15 ----
    const int m = lane & 15;
    float* Pr = P + m * SEQ + hf * (SEQ / 2);
    float mx = -1e30f;
    for (int k = 0; k < SEQ / 2; ++k) mx = fmaxf(mx, Pr[k]);
    mx = fmaxf(mx, __shfl_xor(mx, 16, 32));
    float s = 0.f;
    for (int k = 0; k < SEQ / 2; ++k) { float e = __expf(Pr[k] - mx); Pr[k] = e; s += e; }
    s += __shfl_xor(s, 16, 32);
    const float invr = 1.0f / s;

    // ---- accumulate P/NH into attn_mean (fp32 atomics, L2-resident) -----
    float* amr = attn_mean + (b * SEQ + q0 + m) * SEQ + hf * (SEQ / 2);
    const float w8 = invr * (1.0f / NH);
    for (int k = 0; k < SEQ / 2; ++k) atomicAdd(&amr[k], Pr[k] * w8);
    __syncthreads();

    // ---- attended = P @ V  (16 x 96), 1/l folded into A conversion ------
    v8f acc[6] = {};
    for (int ks = 0; ks < SEQ / 32; ++ks) {
        v16h a = load_fragA_f(P + ks * 32, SEQ, invr);
        const _Float16* Vb = vT + (b * HID + h * HD) * SEQ + ks * 32;
#pragma unroll
        for (int t = 0; t < 6; ++t)
            acc[t] = wmma16(a, load_fragB_h(Vb + (t * 16) * SEQ, SEQ), acc[t]);
    }
#pragma unroll
    for (int t = 0; t < 6; ++t)
#pragma unroll
        for (int j = 0; j < 8; ++j)
            ash[(b * SEQ + q0 + j + 8 * hf) * HID + h * HD + t * 16 + nl] =
                (_Float16)acc[t][j];
}

// ---------------- attended_vector = attn_mean @ vector planes --------------
// per plane (b,c): out[q][d] = sum_k am[b][q][k] * vf[b][c][d][k]
__global__ void __launch_bounds__(256) att2_kernel(
    const float* __restrict__ am, const _Float16* __restrict__ vfh,
    _Float16* __restrict__ av) {
    const int plane = blockIdx.y;                     // b*3+c
    const int b     = plane / 3;
    const int wave  = threadIdx.x >> 5;
    const int gw    = blockIdx.x * 8 + wave;          // 512 waves
    const int mt    = gw & 127, nb = gw >> 7;         // nb 0..3
    const int m0    = mt * 16, n0 = nb * 64;
    const float*    Ab = am + (b * SEQ + m0) * SEQ;
    const _Float16* Bb = vfh + plane * (DVC * SEQ);

    v8f acc[4] = {};
    for (int ks = 0; ks < SEQ / 32; ++ks) {
        v16h a = load_fragA_f(Ab + ks * 32, SEQ, 1.0f);
#pragma unroll
        for (int t = 0; t < 4; ++t)
            acc[t] = wmma16(a, load_fragB_h(Bb + (n0 + t * 16) * SEQ + ks * 32, SEQ), acc[t]);
    }
    const int lane = threadIdx.x & 31, hf = lane >> 4, nl = lane & 15;
    _Float16* Ob = av + plane * (SEQ * DVC);
#pragma unroll
    for (int t = 0; t < 4; ++t)
#pragma unroll
        for (int j = 0; j < 8; ++j)
            Ob[(m0 + j + 8 * hf) * DVC + n0 + t * 16 + nl] = (_Float16)acc[t][j];
}

// ---------------- output projection (Wo) -----------------------------------
__global__ void __launch_bounds__(256) gemm_wo_kernel(
    const _Float16* __restrict__ A, const _Float16* __restrict__ W,
    const float* __restrict__ bias, float* __restrict__ out) {
    const int wave = threadIdx.x >> 5;
    const int gw   = blockIdx.x * 8 + wave;
    const int mt   = gw % (TOK / 16), nb = gw / (TOK / 16);
    const int m0   = mt * 16, n0 = nb * 64;
    v8f acc[4] = {};
#pragma unroll 4
    for (int ks = 0; ks < HID / 32; ++ks) {
        v16h a = load_fragA_h(A + m0 * HID + ks * 32, HID);
#pragma unroll
        for (int t = 0; t < 4; ++t)
            acc[t] = wmma16(a, load_fragB_h(W + (n0 + t * 16) * HID + ks * 32, HID), acc[t]);
    }
    const int lane = threadIdx.x & 31, hf = lane >> 4, nl = lane & 15;
#pragma unroll
    for (int t = 0; t < 4; ++t)
#pragma unroll
        for (int j = 0; j < 8; ++j) {
            int m = m0 + j + 8 * hf, n = n0 + t * 16 + nl;
            out[m * HID + n] = acc[t][j] + bias[n];
        }
}

// ---------------- vector output projection (Wov), per plane ---------------
__global__ void __launch_bounds__(256) gemm_wov_kernel(
    const _Float16* __restrict__ av, const _Float16* __restrict__ W,
    float* __restrict__ vp) {
    const int plane = blockIdx.y;
    const int wave  = threadIdx.x >> 5;
    const int gw    = blockIdx.x * 8 + wave;          // 512 waves
    const int mt    = gw & 127, nb = gw >> 7;
    const int m0    = mt * 16, n0 = nb * 64;
    const _Float16* Ab = av + plane * (SEQ * DVC);

    v8f acc[4] = {};
#pragma unroll
    for (int ks = 0; ks < DVC / 32; ++ks) {
        v16h a = load_fragA_h(Ab + m0 * DVC + ks * 32, DVC);
#pragma unroll
        for (int t = 0; t < 4; ++t)
            acc[t] = wmma16(a, load_fragB_h(W + (n0 + t * 16) * DVC + ks * 32, DVC), acc[t]);
    }
    const int lane = threadIdx.x & 31, hf = lane >> 4, nl = lane & 15;
    float* Ob = vp + plane * (SEQ * DVC);
#pragma unroll
    for (int t = 0; t < 4; ++t)
#pragma unroll
        for (int j = 0; j < 8; ++j)
            Ob[(m0 + j + 8 * hf) * DVC + n0 + t * 16 + nl] = acc[t][j];
}

// ---------------- epilogue: residual + LayerNorm ---------------------------
__global__ void __launch_bounds__(256) ln_kernel(
    const float* __restrict__ sp, const float* __restrict__ x,
    const float* __restrict__ w, const float* __restrict__ bb,
    float* __restrict__ out) {
    const int row = blockIdx.x, t = threadIdx.x;
    const float* pr = sp + row * HID;
    const float* xr = x + row * HID;
    float v[3], s = 0.f, s2 = 0.f;
#pragma unroll
    for (int i = 0; i < 3; ++i) {
        int c = t + i * 256;
        v[i] = pr[c] + xr[c];
        s += v[i]; s2 += v[i] * v[i];
    }
    __shared__ float sh1[256], sh2[256];
    sh1[t] = s; sh2[t] = s2;
    __syncthreads();
    for (int o = 128; o > 0; o >>= 1) {
        if (t < o) { sh1[t] += sh1[t + o]; sh2[t] += sh2[t + o]; }
        __syncthreads();
    }
    float mean = sh1[0] * (1.0f / HID);
    float var  = sh2[0] * (1.0f / HID) - mean * mean;
    float rs   = rsqrtf(var + 1e-5f);
#pragma unroll
    for (int i = 0; i < 3; ++i) {
        int c = t + i * 256;
        out[row * HID + c] = (v[i] - mean) * rs * w[c] + bb[c];
    }
}

// ---------------- epilogue: residual + vector norm -------------------------
__global__ void __launch_bounds__(256) vln_kernel(
    const float* __restrict__ vp, const float* __restrict__ vf,
    const float* __restrict__ w, float* __restrict__ out) {
    const int idx = blockIdx.x * 256 + threadIdx.x;   // < B*S*DVC
    const int d = idx & (DVC - 1);
    const int s = (idx >> 8) & (SEQ - 1);
    const int b = idx >> 19;
    float a0 = vp[((b * 3 + 0) * SEQ + s) * DVC + d] + vf[idx * 3 + 0];
    float a1 = vp[((b * 3 + 1) * SEQ + s) * DVC + d] + vf[idx * 3 + 1];
    float a2 = vp[((b * 3 + 2) * SEQ + s) * DVC + d] + vf[idx * 3 + 2];
    float nrm = sqrtf(a0 * a0 + a1 * a1 + a2 * a2);
    float sc  = w[d] / (nrm + 1e-8f);
    out[idx * 3 + 0] = a0 * sc;
    out[idx * 3 + 1] = a1 * sc;
    out[idx * 3 + 2] = a2 * sc;
}

// ---------------------------------------------------------------------------
extern "C" void kernel_launch(void* const* d_in, const int* in_sizes, int n_in,
                              void* d_out, int out_size, void* d_ws, size_t ws_size,
                              hipStream_t stream) {
    (void)in_sizes; (void)n_in; (void)out_size; (void)ws_size;
    const float* x    = (const float*)d_in[0];
    const float* vf   = (const float*)d_in[1];
    const float* Wq   = (const float*)d_in[2];
    const float* bq   = (const float*)d_in[3];
    const float* Wk   = (const float*)d_in[4];
    const float* bk   = (const float*)d_in[5];
    const float* Wv   = (const float*)d_in[6];
    const float* bv   = (const float*)d_in[7];
    const float* Wo   = (const float*)d_in[8];
    const float* bo   = (const float*)d_in[9];
    const float* Wov  = (const float*)d_in[10];
    const float* lnw  = (const float*)d_in[11];
    const float* lnb  = (const float*)d_in[12];
    const float* vlnw = (const float*)d_in[13];
    // d_in[14] = edge_index: unused by the reference.
    float* out = (float*)d_out;

    char* ws = (char*)d_ws;
    size_t off = 0;
    auto alloc = [&](size_t bytes) {
        void* p = ws + off;
        off += (bytes + 255) & ~(size_t)255;
        return p;
    };
    _Float16* xh   = (_Float16*)alloc((size_t)TOK * HID * 2);   // reused as ash
    _Float16* wqh  = (_Float16*)alloc((size_t)HID * HID * 2);
    _Float16* wkh  = (_Float16*)alloc((size_t)HID * HID * 2);
    _Float16* wvh  = (_Float16*)alloc((size_t)HID * HID * 2);
    _Float16* woh  = (_Float16*)alloc((size_t)HID * HID * 2);
    _Float16* wovh = (_Float16*)alloc((size_t)DVC * DVC * 2);
    _Float16* vfh  = (_Float16*)alloc((size_t)BATCH * 3 * DVC * SEQ * 2);
    _Float16* qh   = (_Float16*)alloc((size_t)TOK * HID * 2);   // reused as av
    _Float16* kh   = (_Float16*)alloc((size_t)TOK * HID * 2);
    _Float16* vT   = (_Float16*)alloc((size_t)TOK * HID * 2);
    float*    am   = (float*)alloc((size_t)BATCH * SEQ * SEQ * 4);
    _Float16* ash  = xh;                                  // after gemm_qkv consumes xh
    _Float16* av   = qh;                                  // after attn1 consumes qh
    float*    sp   = am;                                  // after att2 consumes am
    float*    vp   = (float*)((char*)am + (size_t)TOK * HID * 4);

    const int NE_X  = TOK * HID;       // 3,145,728
    const int NE_W  = HID * HID;       // 589,824
    const int NE_WV = DVC * DVC;       // 65,536
    const int NE_VF = BATCH * SEQ * DVC * 3;

    hipMemsetAsync(am, 0, (size_t)BATCH * SEQ * SEQ * 4, stream);

    cvt_f32_f16_kernel<<<(NE_X + 255) / 256, 256, 0, stream>>>(x, xh, NE_X);
    cvt_f32_f16_kernel<<<(NE_W + 255) / 256, 256, 0, stream>>>(Wq, wqh, NE_W);
    cvt_f32_f16_kernel<<<(NE_W + 255) / 256, 256, 0, stream>>>(Wk, wkh, NE_W);
    cvt_f32_f16_kernel<<<(NE_W + 255) / 256, 256, 0, stream>>>(Wv, wvh, NE_W);
    cvt_f32_f16_kernel<<<(NE_W + 255) / 256, 256, 0, stream>>>(Wo, woh, NE_W);
    cvt_f32_f16_kernel<<<(NE_WV + 255) / 256, 256, 0, stream>>>(Wov, wovh, NE_WV);
    cvt_vecf_kernel<<<(NE_VF + 255) / 256, 256, 0, stream>>>(vf, vfh);

    gemm_qkv_kernel<<<dim3((TOK / 16) * (HID / 64) / 8, 1, 3), 256, 0, stream>>>(
        xh, wqh, wkh, wvh, bq, bk, bv, qh, kh, vT);

    attn1_kernel<<<dim3(BATCH * NH * (SEQ / 16)), dim3(32), 16 * SEQ * 4, stream>>>(
        qh, kh, vT, am, ash);

    att2_kernel<<<dim3((SEQ / 16) * (DVC / 64) / 8, BATCH * 3), 256, 0, stream>>>(
        am, vfh, av);

    gemm_wo_kernel<<<dim3((TOK / 16) * (HID / 64) / 8), 256, 0, stream>>>(
        ash, woh, bo, sp);

    gemm_wov_kernel<<<dim3((SEQ / 16) * (DVC / 64) / 8, BATCH * 3), 256, 0, stream>>>(
        av, wovh, vp);

    ln_kernel<<<TOK, 256, 0, stream>>>(sp, x, lnw, lnb, out);
    vln_kernel<<<(BATCH * SEQ * DVC) / 256, 256, 0, stream>>>(
        vp, vf, vlnw, out + (size_t)TOK * HID);
}
